// SelfAttention_13460427505889
// MI455X (gfx1250) — matrix operand order repaired and checked by reading, not compiled, not run on hardware
//
#include <hip/hip_runtime.h>
#include <hip/hip_bf16.h>
#include <math.h>

typedef _Float16 half_t;
typedef __attribute__((ext_vector_type(16))) _Float16 v16h;
typedef __attribute__((ext_vector_type(8)))  _Float16 v8h;
typedef __attribute__((ext_vector_type(4)))  _Float16 v4h;
typedef __attribute__((ext_vector_type(8)))  float    v8f;

static constexpr int Bn = 2;
static constexpr int Tn = 2048;
static constexpr int Cn = 2048;
static constexpr int Hn = 16;
static constexpr int KVn = 4;
static constexpr int Dn = 128;

#if defined(__AMDGCN__) && __has_builtin(__builtin_amdgcn_tensor_load_to_lds)
#define USE_TDM 1
#else
#define USE_TDM 0
#endif

#if USE_TDM
typedef __attribute__((ext_vector_type(4))) unsigned int u32x4;
typedef __attribute__((ext_vector_type(4))) int          i32x4;
typedef __attribute__((ext_vector_type(8))) int          i32x8;

__device__ inline unsigned lds_addr_of(const void* p) {
  // flat LDS address: high word is the shared aperture, low 32 bits = LDS byte offset
  return (unsigned)(uintptr_t)p;
}

// 2D tile DMA: global (row-major, stride0 elems of 2B) -> LDS with row padding.
// pad encoding: row payload = (2<<pad_interval) DWORDs, pad = (pad_amount+1) DWORDs.
__device__ inline void tdm_load_2d(unsigned lds_addr, const void* gptr,
                                   unsigned tensor_d0, unsigned tensor_d1,
                                   unsigned long long stride0,
                                   unsigned tile_d0, unsigned tile_d1,
                                   unsigned pad_interval, unsigned pad_amount) {
  unsigned long long ga = (unsigned long long)(uintptr_t)gptr;
  u32x4 g0;
  g0[0] = 1u;                                        // count=1, user descriptor
  g0[1] = lds_addr;                                  // LDS byte address
  g0[2] = (unsigned)(ga & 0xffffffffull);            // global_addr[31:0]
  g0[3] = (unsigned)((ga >> 32) & 0x1ffffffull) | (2u << 30);  // [56:32] | type=2
  i32x8 g1;
  g1[0] = (int)((1u << 16)                           // data_size = 2 bytes
                | (1u << 20)                         // pad_enable
                | (pad_interval << 22) | (pad_amount << 25));
  g1[1] = (int)((tensor_d0 & 0xffffu) << 16);        // barrier_addr=0 | dim0 lo
  g1[2] = (int)(((tensor_d0 >> 16) & 0xffffu) | ((tensor_d1 & 0xffffu) << 16));
  g1[3] = (int)(((tensor_d1 >> 16) & 0xffffu) | (tile_d0 << 16));
  g1[4] = (int)(tile_d1 & 0xffffu);                  // tile_dim2 = 0 (2D)
  g1[5] = (int)(unsigned)(stride0 & 0xffffffffull);
  g1[6] = (int)(unsigned)((stride0 >> 32) & 0xffffull);  // dim1_stride unused
  g1[7] = 0;
  i32x4 z4 = {0, 0, 0, 0};
#if defined(__clang_major__) && __clang_major__ >= 23
  i32x8 z8 = {0, 0, 0, 0, 0, 0, 0, 0};
  __builtin_amdgcn_tensor_load_to_lds(g0, g1, z4, z4, z8, 0);
#else
  __builtin_amdgcn_tensor_load_to_lds(g0, g1, z4, z4, 0);
#endif
}
#endif  // USE_TDM

__device__ inline void tdm_fence(int wave) {
#if USE_TDM
  if (wave == 0) __builtin_amdgcn_s_wait_tensorcnt(0);
#else
  (void)wave;
#endif
}

__device__ inline v8f zero8() {
  v8f z = {0.f, 0.f, 0.f, 0.f, 0.f, 0.f, 0.f, 0.f};
  return z;
}

__device__ inline v8f wmma16(v16h a, v16h b, v8f c) {
  return __builtin_amdgcn_wmma_f32_16x16x32_f16(false, a, false, b, (short)0, c,
                                                false, false);
}

// A-fragment (16x32 f16): two contiguous 16B chunks at k0+g*8, +16.
__device__ inline v16h ldfragA(const half_t* t, int row0, int k0, int stride) {
  const int lane = threadIdx.x & 31;
  const int g = lane >> 4;
  const half_t* p = t + (size_t)(row0 + (lane & 15)) * stride + k0 + g * 8;
  v8h lo = *(const v8h*)(p);
  v8h hi = *(const v8h*)(p + 16);
  v16h r;
#pragma unroll
  for (int i = 0; i < 8; ++i) { r[i] = lo[i]; r[i + 8] = hi[i]; }
  return r;
}

// B-fragment (32x16 f16) from an N-major (N x K) tile: chunks at k0+g*8, +16.
__device__ inline v16h ldfragB(const half_t* t, int row0, int k0, int stride) {
  const int lane = threadIdx.x & 31;
  const int g = lane >> 4;
  const half_t* p = t + (size_t)(row0 + (lane & 15)) * stride + k0 + g * 8;
  v8h lo = *(const v8h*)(p);
  v8h hi = *(const v8h*)(p + 16);
  v16h r;
#pragma unroll
  for (int i = 0; i < 8; ++i) { r[i] = lo[i]; r[i + 8] = hi[i]; }
  return r;
}

// ---------------- elementwise converters ----------------

__global__ __launch_bounds__(256) void cvt_f32_f16x4(const float* __restrict__ s,
                                                     half_t* __restrict__ d, int n4) {
  int i = blockIdx.x * blockDim.x + threadIdx.x;
  if (i >= n4) return;
  float4 v = ((const float4*)s)[i];
  v4h o;
  o[0] = (half_t)v.x; o[1] = (half_t)v.y; o[2] = (half_t)v.z; o[3] = (half_t)v.w;
  ((v4h*)d)[i] = o;
}

// W (Cn x N) row-major f32 -> Wt (N x Cn) row-major f16.
__global__ __launch_bounds__(256) void wtrans(const float* __restrict__ W,
                                              half_t* __restrict__ Wt, int N) {
  int idx = blockIdx.x * blockDim.x + threadIdx.x;
  int k = idx & (Cn - 1);
  int n = idx >> 11;
  if (n >= N) return;
  Wt[(size_t)n * Cn + k] = (half_t)W[(size_t)k * N + n];
}

// v f32 [(b*T+t), kv*D+d] -> vt f16 [b][kv][d][t]
__global__ __launch_bounds__(256) void vtransk(const float* __restrict__ vf,
                                               half_t* __restrict__ vt) {
  int idx = blockIdx.x * blockDim.x + threadIdx.x;
  int t = idx & (Tn - 1);
  int d = (idx >> 11) & (Dn - 1);
  int kv = (idx >> 18) & (KVn - 1);
  int b = idx >> 20;
  vt[idx] = (half_t)vf[((size_t)(b * Tn + t)) * (KVn * Dn) + kv * Dn + d];
}

// ---------------- fused RoPE + RMSNorm ----------------
__global__ __launch_bounds__(128) void rope_rms(const float* __restrict__ src,
                                                const float* __restrict__ cosp,
                                                const float* __restrict__ sinp,
                                                half_t* __restrict__ dst, int nh) {
  const int wave = threadIdx.x >> 5;
  const int lane = threadIdx.x & 31;
  const int row = blockIdx.x * 4 + wave;
  const int total = Bn * Tn * nh;
  if (row >= total) return;
  const int b = row / (Tn * nh);
  const int rem = row - b * Tn * nh;
  const int t = rem / nh;
  const int h = rem - t * nh;
  const float* p = src + ((size_t)(b * Tn + t) * nh + h) * Dn;

  float o[4];
  float ss = 0.f;
#pragma unroll
  for (int j = 0; j < 2; ++j) {
    int i = lane + j * 32;
    float t1 = p[i];
    float t2 = p[i + 64];
    float c = cosp[t * 64 + i];
    float s = sinp[t * 64 + i];
    float a = t1 * c + t2 * s;
    float bb = -t1 * s + t2 * c;
    o[j * 2] = a;
    o[j * 2 + 1] = bb;
    ss += a * a + bb * bb;
  }
#pragma unroll
  for (int off = 1; off < 32; off <<= 1) ss += __shfl_xor(ss, off);
  float inv = rsqrtf(ss * (1.0f / 128.0f) + 1.1920929e-07f);

  half_t* q = dst + (((size_t)b * nh + h) * Tn + t) * Dn;
#pragma unroll
  for (int j = 0; j < 2; ++j) {
    int i = lane + j * 32;
    q[i] = (half_t)(o[j * 2] * inv);
    q[i + 64] = (half_t)(o[j * 2 + 1] * inv);
  }
}

// -------- WMMA GEMM: C(MxN,f32) = A(MxK,f16) * Bt(NxK,f16)^T, BK=64, 2-buf --------
__global__ __launch_bounds__(256) void gemm_wmma(const half_t* __restrict__ A,
                                                 const half_t* __restrict__ Bt,
                                                 float* __restrict__ Cm,
                                                 int M, int N, int K) {
  constexpr int LD = 72;                     // 64 + 8 pad halves
  constexpr int TILE = 128 * LD;             // halves per (A or B) tile
  __shared__ half_t smem[2 * 2 * TILE];      // [buf][A,B]  = 73728 B
  const int tid = threadIdx.x;
  const int lane = tid & 31;
  const int wave = tid >> 5;
  const int wm = wave >> 1;
  const int wn = wave & 1;
  const int m0 = blockIdx.y * 128;
  const int n0 = blockIdx.x * 128;

  auto issue = [&](int k0, int bufi) {
    half_t* sA = smem + bufi * 2 * TILE;
    half_t* sB = sA + TILE;
#if USE_TDM
    if (wave == 0) {
      // rows of 64 elems (128B = 32 DW -> pad_interval 4), pad 16B (4 DW -> 3)
      tdm_load_2d(lds_addr_of(sA), A + (size_t)m0 * K + k0,
                  (unsigned)K, (unsigned)M, (unsigned long long)K, 64, 128, 4, 3);
      tdm_load_2d(lds_addr_of(sB), Bt + (size_t)n0 * K + k0,
                  (unsigned)K, (unsigned)N, (unsigned long long)K, 64, 128, 4, 3);
    }
#else
#pragma unroll
    for (int i = 0; i < 4; ++i) {            // 1024 x 16B chunks, 4/thread
      int c = tid + i * 256;
      int row = c >> 3;
      int col = c & 7;
      *(v8h*)&sA[row * LD + col * 8] =
          *(const v8h*)&A[(size_t)(m0 + row) * K + k0 + col * 8];
      *(v8h*)&sB[row * LD + col * 8] =
          *(const v8h*)&Bt[(size_t)(n0 + row) * K + k0 + col * 8];
    }
#endif
  };

  v8f acc[2][4];
#pragma unroll
  for (int a = 0; a < 2; ++a)
#pragma unroll
    for (int b = 0; b < 4; ++b) acc[a][b] = zero8();

  int buf = 0;
  issue(0, 0);
  tdm_fence(wave);
  __syncthreads();

  for (int k0 = 0; k0 < K; k0 += 64) {
    if (k0 + 64 < K) issue(k0 + 64, buf ^ 1);
    const half_t* sA = smem + buf * 2 * TILE;
    const half_t* sB = sA + TILE;
    v16h af[2][2];
#pragma unroll
    for (int ks = 0; ks < 2; ++ks) {
      af[0][ks] = ldfragA(sA, wm * 32, ks * 32, LD);
      af[1][ks] = ldfragA(sA, wm * 32 + 16, ks * 32, LD);
    }
#pragma unroll
    for (int b = 0; b < 4; ++b)
#pragma unroll
      for (int ks = 0; ks < 2; ++ks) {
        v16h bf = ldfragB(sB, wn * 64 + b * 16, ks * 32, LD);
        acc[0][b] = wmma16(af[0][ks], bf, acc[0][b]);
        acc[1][b] = wmma16(af[1][ks], bf, acc[1][b]);
      }
    tdm_fence(wave);
    __syncthreads();
    buf ^= 1;
  }

  const int g2 = lane >> 4;
#pragma unroll
  for (int a = 0; a < 2; ++a)
#pragma unroll
    for (int b = 0; b < 4; ++b)
#pragma unroll
      for (int r = 0; r < 8; ++r)
        Cm[(size_t)(m0 + wm * 32 + a * 16 + r + 8 * g2) * N + n0 + wn * 64 +
           b * 16 + (lane & 15)] = acc[a][b][r];
}

// ---------------- flash attention (double-buffered K/V tiles) ----------------
__global__ __launch_bounds__(128) void attn_wmma(const half_t* __restrict__ qh,
                                                 const half_t* __restrict__ kh,
                                                 const half_t* __restrict__ vt,
                                                 half_t* __restrict__ yh) {
  constexpr int LK = 136;                    // 128 + 8 pad
  constexpr int LV = 72;                     // 64 + 8 pad
  constexpr int KVTILE = 64 * LK + 128 * LV; // halves per buffer
  __shared__ half_t sKV[2 * KVTILE];         // 71680 B
  __shared__ half_t sP[4 * 16 * LV];         // per-wave P staging

  const int tid = threadIdx.x;
  const int lane = tid & 31;
  const int wave = tid >> 5;
  const int g2 = lane >> 4;
  const int qb = blockIdx.x;
  const int h = blockIdx.y;
  const int b = blockIdx.z;
  const int kv = h >> 2;
  const int qbase = qb * 64;
  const int kbLast = qb;

  const half_t* qtile = qh + (((size_t)b * Hn + h) * Tn + qbase + wave * 16) * Dn;
  const half_t* kbase = kh + (((size_t)b * KVn + kv) * Tn) * Dn;
  const half_t* vbase = vt + (((size_t)b * KVn + kv) * Dn) * Tn;
  half_t* pw = sP + wave * 16 * LV;

  auto issue = [&](int kb, int bufi) {
    half_t* kbuf = sKV + bufi * KVTILE;
    half_t* vbuf = kbuf + 64 * LK;
#if USE_TDM
    if (wave == 0) {
      // K tile: 64 rows x 128 elems (256B = 64 DW -> interval 5), pad 16B
      tdm_load_2d(lds_addr_of(kbuf), kbase + (size_t)kb * 64 * Dn,
                  Dn, Tn, Dn, Dn, 64, 5, 3);
      // V tile: 128 rows x 64 elems (128B = 32 DW -> interval 4), pad 16B
      tdm_load_2d(lds_addr_of(vbuf), vbase + (size_t)kb * 64,
                  Tn, Dn, Tn, 64, 128, 4, 3);
    }
#else
#pragma unroll
    for (int i = 0; i < 8; ++i) {
      int c = tid + i * 128;
      int krow = c >> 4, kcol = c & 15;
      *(v8h*)&kbuf[krow * LK + kcol * 8] =
          *(const v8h*)&kbase[(size_t)(kb * 64 + krow) * Dn + kcol * 8];
      int vrow = c >> 3, vcol = c & 7;
      *(v8h*)&vbuf[vrow * LV + vcol * 8] =
          *(const v8h*)&vbase[(size_t)vrow * Tn + kb * 64 + vcol * 8];
    }
#endif
  };

  v16h qf[4];
#pragma unroll
  for (int ks = 0; ks < 4; ++ks) qf[ks] = ldfragA(qtile, 0, ks * 32, Dn);

  v8f accO[8];
#pragma unroll
  for (int dt = 0; dt < 8; ++dt) accO[dt] = zero8();
  float m2[8], lsum[8];
#pragma unroll
  for (int r = 0; r < 8; ++r) { m2[r] = -1e30f; lsum[r] = 0.f; }

  const float sc2 = 0.088388347648318447f * 1.4426950408889634f; // log2e/sqrt(D)

  int buf = 0;
  issue(0, 0);
  tdm_fence(wave);
  __syncthreads();

  for (int kb = 0; kb <= kbLast; ++kb) {
    if (kb < kbLast) issue(kb + 1, buf ^ 1);
    const half_t* kbuf = sKV + buf * KVTILE;
    const half_t* vbuf = kbuf + 64 * LK;

    // S = Q x K^T  (16 x 64 per wave)
    v8f s[4];
#pragma unroll
    for (int nt = 0; nt < 4; ++nt) {
      s[nt] = zero8();
#pragma unroll
      for (int ks = 0; ks < 4; ++ks)
        s[nt] = wmma16(qf[ks], ldfragB(kbuf, nt * 16, ks * 32, LK), s[nt]);
    }

    const bool diag = (kb == kbLast);
    float pv[4][8];
#pragma unroll
    for (int r = 0; r < 8; ++r) {
      const int trow = qbase + wave * 16 + r + 8 * g2;
      float mx = -1e30f;
#pragma unroll
      for (int nt = 0; nt < 4; ++nt) {
        float v = s[nt][r] * sc2;
        if (diag && (kb * 64 + nt * 16 + (lane & 15)) > trow) v = -1e30f;
        pv[nt][r] = v;
        mx = fmaxf(mx, v);
      }
#pragma unroll
      for (int off = 1; off < 16; off <<= 1) mx = fmaxf(mx, __shfl_xor(mx, off));
      float mn = fmaxf(m2[r], mx);
      float alpha = exp2f(m2[r] - mn);
      m2[r] = mn;
      float rs = 0.f;
#pragma unroll
      for (int nt = 0; nt < 4; ++nt) {
        float p = exp2f(pv[nt][r] - mn);
        pv[nt][r] = p;
        rs += p;
      }
#pragma unroll
      for (int off = 1; off < 16; off <<= 1) rs += __shfl_xor(rs, off);
      lsum[r] = lsum[r] * alpha + rs;
#pragma unroll
      for (int dt = 0; dt < 8; ++dt) accO[dt][r] *= alpha;
    }

    // stage P (same-wave LDS ops are in order)
#pragma unroll
    for (int nt = 0; nt < 4; ++nt)
#pragma unroll
      for (int r = 0; r < 8; ++r)
        pw[(r + 8 * g2) * LV + nt * 16 + (lane & 15)] = (half_t)pv[nt][r];

    // O += P x V  (16 x 128 per wave)
#pragma unroll
    for (int dt = 0; dt < 8; ++dt)
#pragma unroll
      for (int ks = 0; ks < 2; ++ks)
        accO[dt] = wmma16(ldfragA(pw, 0, ks * 32, LV),
                          ldfragB(vbuf, dt * 16, ks * 32, LV), accO[dt]);
    tdm_fence(wave);
    __syncthreads();
    buf ^= 1;
  }

#pragma unroll
  for (int r = 0; r < 8; ++r) {
    const int trow = qbase + wave * 16 + r + 8 * g2;
    const float inv = 1.0f / lsum[r];
    half_t* yrow = yh + ((size_t)b * Tn + trow) * Cn + h * Dn;
#pragma unroll
    for (int dt = 0; dt < 8; ++dt)
      yrow[dt * 16 + (lane & 15)] = (half_t)(accO[dt][r] * inv);
  }
}

// ---------------- launcher ----------------
extern "C" void kernel_launch(void* const* d_in, const int* in_sizes, int n_in,
                              void* d_out, int out_size, void* d_ws, size_t ws_size,
                              hipStream_t stream) {
  (void)in_sizes; (void)n_in; (void)out_size; (void)ws_size;
  const float* x    = (const float*)d_in[0];
  const float* cosp = (const float*)d_in[1];
  const float* sinp = (const float*)d_in[2];
  const float* Wq   = (const float*)d_in[3];
  const float* Wk   = (const float*)d_in[4];
  const float* Wv   = (const float*)d_in[5];
  const float* Wo   = (const float*)d_in[6];
  float* out = (float*)d_out;

  char* ws = (char*)d_ws;
  size_t off = 0;
  auto alloc = [&](size_t bytes) -> void* {
    void* p = ws + off;
    off += (bytes + 255) & ~(size_t)255;
    return p;
  };
  const size_t BT = (size_t)Bn * Tn;
  half_t* xh  = (half_t*)alloc(BT * Cn * 2);
  half_t* WqT = (half_t*)alloc((size_t)Cn * (Hn * Dn) * 2);
  half_t* WkT = (half_t*)alloc((size_t)Cn * (KVn * Dn) * 2);
  half_t* WvT = (half_t*)alloc((size_t)Cn * (KVn * Dn) * 2);
  half_t* WoT = (half_t*)alloc((size_t)Cn * Cn * 2);
  float*  qf  = (float*)alloc(BT * (Hn * Dn) * 4);
  float*  kf  = (float*)alloc(BT * (KVn * Dn) * 4);
  float*  vf  = (float*)alloc(BT * (KVn * Dn) * 4);
  half_t* qhh = (half_t*)alloc(BT * (Hn * Dn) * 2);
  half_t* khh = (half_t*)alloc(BT * (KVn * Dn) * 2);
  half_t* vth = (half_t*)alloc(BT * (KVn * Dn) * 2);
  half_t* yhh = (half_t*)qf;                          // alias: qf dead by then

  cvt_f32_f16x4<<<(BT * Cn / 4 + 255) / 256, 256, 0, stream>>>(x, xh,
      (int)(BT * Cn / 4));
  wtrans<<<(Hn * Dn * Cn) / 256, 256, 0, stream>>>(Wq, WqT, Hn * Dn);
  wtrans<<<(KVn * Dn * Cn) / 256, 256, 0, stream>>>(Wk, WkT, KVn * Dn);
  wtrans<<<(KVn * Dn * Cn) / 256, 256, 0, stream>>>(Wv, WvT, KVn * Dn);
  wtrans<<<(Cn * Cn) / 256, 256, 0, stream>>>(Wo, WoT, Cn);

  gemm_wmma<<<dim3((Hn * Dn) / 128, BT / 128), 256, 0, stream>>>(xh, WqT, qf,
      (int)BT, Hn * Dn, Cn);
  gemm_wmma<<<dim3((KVn * Dn) / 128, BT / 128), 256, 0, stream>>>(xh, WkT, kf,
      (int)BT, KVn * Dn, Cn);
  gemm_wmma<<<dim3((KVn * Dn) / 128, BT / 128), 256, 0, stream>>>(xh, WvT, vf,
      (int)BT, KVn * Dn, Cn);

  rope_rms<<<(Bn * Tn * Hn) / 4, 128, 0, stream>>>(qf, cosp, sinp, qhh, Hn);
  rope_rms<<<(Bn * Tn * KVn) / 4, 128, 0, stream>>>(kf, cosp, sinp, khh, KVn);
  vtransk<<<(Bn * KVn * Dn * Tn) / 256, 256, 0, stream>>>(vf, vth);

  attn_wmma<<<dim3(Tn / 64, Hn, Bn), 128, 0, stream>>>(qhh, khh, vth, yhh);

  gemm_wmma<<<dim3(Cn / 128, BT / 128), 256, 0, stream>>>(yhh, WoT, out,
      (int)BT, Cn, Cn);
}
